// MassiveLSTM_23132693856602
// MI455X (gfx1250) — compile-verified, matrix operand order checked by reading, not run on hardware
//
#include <hip/hip_runtime.h>

// ---------------------------------------------------------------------------
// Types for CDNA5 WMMA (wave32)
// ---------------------------------------------------------------------------
typedef __bf16 bf16_t;
typedef bf16_t v16bf __attribute__((ext_vector_type(16)));
typedef float  v8f   __attribute__((ext_vector_type(8)));

#define LSTM_B  16
#define LSTM_T  256
#define LSTM_IN 512
#define LSTM_H  1024
#define LSTM_G  4096   // 4*H
#define N_LAYERS 8

// ---------------------------------------------------------------------------
// Helpers
// ---------------------------------------------------------------------------
__device__ __forceinline__ unsigned short f2bf(float f) {
    unsigned u = __float_as_uint(f);
    u += 0x7FFFu + ((u >> 16) & 1u);          // round-to-nearest-even
    return (unsigned short)(u >> 16);
}
__device__ __forceinline__ float bf2f(unsigned short s) {
    return __uint_as_float((unsigned)s << 16);
}
__device__ __forceinline__ float sigmoidf(float x) {
    return 1.0f / (1.0f + __expf(-x));
}

__device__ __forceinline__ v8f wmma_bf16(v16bf a, v16bf b, v8f c) {
    return __builtin_amdgcn_wmma_f32_16x16x32_bf16(
        /*neg_a=*/false, a, /*neg_b=*/false, b,
        /*c_mod=*/(short)0, c, /*reuse_a=*/false, /*reuse_b=*/false);
}

// Load a 16x32 bf16 tile in the ISA A/B VGPR layout.
// base points at the tile's top-left element of a row-major (16 x ld) region.
// Lane L (L<16):   row L,    K = {kbase+0..7,  kbase+16..23}
// Lane L (L>=16):  row L-16, K = {kbase+8..15, kbase+24..31}
__device__ __forceinline__ v16bf load_frag(const unsigned short* base, int ld,
                                           int kbase, int lane) {
    const int hsel = lane >> 4;
    const int row  = lane & 15;
    const unsigned short* p = base + (size_t)row * ld + kbase + hsel * 8;
    union { v16bf v; uint4 q[2]; } u;
    u.q[0] = *reinterpret_cast<const uint4*>(p);
    u.q[1] = *reinterpret_cast<const uint4*>(p + 16);
    return u.v;
}

// ---------------------------------------------------------------------------
// f32 -> bf16 conversion (grid-stride)
// ---------------------------------------------------------------------------
__global__ void k_f32_to_bf16(const float* __restrict__ src,
                              unsigned short* __restrict__ dst, int n) {
    int i = blockIdx.x * blockDim.x + threadIdx.x;
    int stride = gridDim.x * blockDim.x;
    for (; i < n; i += stride) dst[i] = f2bf(src[i]);
}

// ---------------------------------------------------------------------------
// Input projection GEMM: xg[m,n] = sum_k A[m,k]*W[n,k] + bih[n] + bhh[n]
// A: (M x K) bf16 row-major, W: (N x K) bf16 row-major, out f32 (M x N).
// One wave per 32x64 output tile (2 M-tiles x 4 N-tiles): per K-step it
// loads 2 A frags + 4 B frags and issues 8 wmma_f32_16x16x32_bf16
// (0.75 fragment loads per WMMA; 8 independent accumulate chains).
// __launch_bounds__(256,1): full VGPR budget for the register tile, no spill.
// ---------------------------------------------------------------------------
__global__ void __launch_bounds__(256, 1)
k_gemm_xg(const unsigned short* __restrict__ A,
          const unsigned short* __restrict__ W,
          const float* __restrict__ bih,
          const float* __restrict__ bhh,
          float* __restrict__ out,
          int M, int N, int K) {
    const int lane = threadIdx.x & 31;
    const int wave = threadIdx.x >> 5;
    const int tile = blockIdx.x * (blockDim.x >> 5) + wave;
    const int ntiles = N >> 6;                 // 64-wide N blocks
    const int mt = tile / ntiles;              // 32-high M block
    const int nt = tile % ntiles;
    if (mt >= (M >> 5)) return;

    const unsigned short* a0 = A + (size_t)(mt * 32) * K;
    const unsigned short* w0 = W + (size_t)(nt * 64) * K;

    v8f acc[2][4];
#pragma unroll
    for (int i = 0; i < 2; ++i)
#pragma unroll
        for (int j = 0; j < 4; ++j) acc[i][j] = (v8f){};

    for (int k = 0; k < K; k += 32) {
        v16bf af[2], bf[4];
#pragma unroll
        for (int i = 0; i < 2; ++i)
            af[i] = load_frag(a0 + (size_t)(i * 16) * K, K, k, lane);
#pragma unroll
        for (int j = 0; j < 4; ++j)
            bf[j] = load_frag(w0 + (size_t)(j * 16) * K, K, k, lane);
#pragma unroll
        for (int i = 0; i < 2; ++i)
#pragma unroll
            for (int j = 0; j < 4; ++j)
                acc[i][j] = wmma_bf16(af[i], bf[j], acc[i][j]);
    }

    const int hsel = lane >> 4;
#pragma unroll
    for (int j = 0; j < 4; ++j) {
        const int n = nt * 64 + j * 16 + (lane & 15);
        const float bb = bih[n] + bhh[n];
#pragma unroll
        for (int i = 0; i < 2; ++i) {
            float* ocol = out + (size_t)(mt * 32 + i * 16) * N + n;
#pragma unroll
            for (int r = 0; r < 8; ++r) {
                int m = r + 8 * hsel;          // C layout: VGPR r -> row r+8*(lane>=16)
                ocol[(size_t)m * N] = acc[i][j][r] + bb;
            }
        }
    }
}

// ---------------------------------------------------------------------------
// Per-layer state init: zero h double-buffers, c state, grid-barrier counter
// ---------------------------------------------------------------------------
__global__ void k_init_state(unsigned short* __restrict__ h0,
                             unsigned short* __restrict__ h1,
                             float* __restrict__ c,
                             unsigned* __restrict__ cnt) {
    int i = blockIdx.x * blockDim.x + threadIdx.x;
    if (i == 0) *cnt = 0u;
    int stride = gridDim.x * blockDim.x;
    for (int j = i; j < LSTM_B * LSTM_H; j += stride) {
        h0[j] = 0; h1[j] = 0; c[j] = 0.0f;
    }
}

// ---------------------------------------------------------------------------
// Persistent recurrent kernel (one layer, all T steps).
// Grid: 8 blocks x 256 threads = 64 waves; wave g owns h-columns [16g,16g+16).
// Per step: async-copy h_{t-1} (16x1024 bf16, 32KB) global->LDS with
// GLOBAL_LOAD_ASYNC_TO_LDS_B128 (ASYNCcnt path, no VGPR round-trip), then
// 4 gate tiles (i,f,g,o) x K=1024 -> 128 wmma per wave per step.
// Whh streamed from the 192MB L2. h-state double-buffered; one accumulating
// atomic grid barrier per step with s_sleep backoff.
// ---------------------------------------------------------------------------
__global__ void __launch_bounds__(256, 1)
k_lstm_recurrent(const unsigned short* __restrict__ whh, // (G x H) bf16
                 const float* __restrict__ xg,           // (B*T x G) f32
                 unsigned short* __restrict__ seq_out,   // (B*T x H) bf16
                 unsigned short* __restrict__ h0,
                 unsigned short* __restrict__ h1,
                 float* __restrict__ c_state,            // (B x H) f32
                 unsigned* __restrict__ barrier_cnt) {
    __shared__ unsigned short hsh[LSTM_B * LSTM_H];   // 32 KB

    const int tid   = threadIdx.x;
    const int lane  = tid & 31;
    const int hsel  = lane >> 4;
    const int gwave = blockIdx.x * (blockDim.x >> 5) + (tid >> 5); // 0..63
    const int c0    = gwave << 4;                 // base h column of this wave
    const int col   = c0 + (lane & 15);
    const unsigned nwg = gridDim.x;

    // Gate weight panels for this wave (fixed for the whole layer).
    const unsigned short* wq0 = whh + (size_t)(0 * LSTM_H + c0) * LSTM_H;
    const unsigned short* wq1 = whh + (size_t)(1 * LSTM_H + c0) * LSTM_H;
    const unsigned short* wq2 = whh + (size_t)(2 * LSTM_H + c0) * LSTM_H;
    const unsigned short* wq3 = whh + (size_t)(3 * LSTM_H + c0) * LSTM_H;

    // Wave-relative LDS byte address of this thread's staging slot
    // (low 32 bits of a generic shared pointer == LDS byte offset).
    const unsigned lds_base =
        (unsigned)(uintptr_t)(&hsh[0]) + (unsigned)tid * 128u;

    for (int t = 0; t < LSTM_T; ++t) {
        const unsigned short* hprev = (t & 1) ? h1 : h0;
        unsigned short*       hnext = (t & 1) ? h0 : h1;

        // Stage h_{t-1} into LDS: 8 async b128 beats per thread (128 B each).
        {
            const char* gsrc = (const char*)hprev + (size_t)tid * 128u;
            asm volatile(
                "global_load_async_to_lds_b128 %0, %1, off\n\t"
                "global_load_async_to_lds_b128 %0, %1, off offset:16\n\t"
                "global_load_async_to_lds_b128 %0, %1, off offset:32\n\t"
                "global_load_async_to_lds_b128 %0, %1, off offset:48\n\t"
                "global_load_async_to_lds_b128 %0, %1, off offset:64\n\t"
                "global_load_async_to_lds_b128 %0, %1, off offset:80\n\t"
                "global_load_async_to_lds_b128 %0, %1, off offset:96\n\t"
                "global_load_async_to_lds_b128 %0, %1, off offset:112\n\t"
                "s_wait_asynccnt 0x0"
                :: "v"(lds_base), "v"(gsrc) : "memory");
        }
        __syncthreads();

        // Accumulators seeded with xg (already includes bih + bhh).
        v8f acc[4];
#pragma unroll
        for (int q = 0; q < 4; ++q) {
            v8f a = {};
#pragma unroll
            for (int r = 0; r < 8; ++r) {
                int b = r + 8 * hsel;
                a[r] = xg[(size_t)(b * LSTM_T + t) * LSTM_G + q * LSTM_H + col];
            }
            acc[q] = a;
        }

        // Recurrent GEMM: g += h_{t-1} @ Whh^T  (A from LDS, B from L2)
        for (int k = 0; k < LSTM_H; k += 32) {
            v16bf af = load_frag(hsh, LSTM_H, k, lane);
            acc[0] = wmma_bf16(af, load_frag(wq0, LSTM_H, k, lane), acc[0]);
            acc[1] = wmma_bf16(af, load_frag(wq1, LSTM_H, k, lane), acc[1]);
            acc[2] = wmma_bf16(af, load_frag(wq2, LSTM_H, k, lane), acc[2]);
            acc[3] = wmma_bf16(af, load_frag(wq3, LSTM_H, k, lane), acc[3]);
        }

        // Gates + state update (each lane owns 8 (batch,col) cells).
#pragma unroll
        for (int r = 0; r < 8; ++r) {
            int b = r + 8 * hsel;
            float vi = acc[0][r], vf = acc[1][r], vg = acc[2][r], vo = acc[3][r];
            int ci = b * LSTM_H + col;
            float cp = c_state[ci];
            float cn = sigmoidf(vf) * cp + sigmoidf(vi) * tanhf(vg);
            float hv = sigmoidf(vo) * tanhf(cn);
            c_state[ci] = cn;
            unsigned short hb = f2bf(hv);
            hnext[ci] = hb;
            seq_out[(size_t)(b * LSTM_T + t) * LSTM_H + col] = hb;
        }

        // Grid-wide barrier (accumulating counter, sense-free).
        __threadfence();
        __syncthreads();
        if (tid == 0) {
            atomicAdd(barrier_cnt, 1u);
            unsigned target = nwg * (unsigned)(t + 1);
            while (__hip_atomic_load(barrier_cnt, __ATOMIC_ACQUIRE,
                                     __HIP_MEMORY_SCOPE_AGENT) < target) {
                __builtin_amdgcn_s_sleep(1);
            }
        }
        __syncthreads();
    }
}

// ---------------------------------------------------------------------------
// FC head (tiny: ~21 MFLOP total)
// ---------------------------------------------------------------------------
__global__ void k_fc1(const unsigned short* __restrict__ seq, // last layer (B*T x H) bf16
                      const float* __restrict__ w,            // (512 x 1024)
                      const float* __restrict__ bias,
                      float* __restrict__ out) {              // (16 x 512)
    int i = blockIdx.x * blockDim.x + threadIdx.x;
    if (i >= LSTM_B * 512) return;
    int m = i >> 9, n = i & 511;
    const unsigned short* hrow = seq + (size_t)(m * LSTM_T + (LSTM_T - 1)) * LSTM_H;
    const float* wrow = w + (size_t)n * LSTM_H;
    float s = bias[n];
    for (int k = 0; k < LSTM_H; ++k) s += bf2f(hrow[k]) * wrow[k];
    out[i] = s > 0.0f ? s : 0.0f;
}

__global__ void k_fc2(const float* __restrict__ in,   // (16 x 512)
                      const float* __restrict__ w,    // (256 x 512)
                      const float* __restrict__ bias,
                      float* __restrict__ out) {      // (16 x 256)
    int i = blockIdx.x * blockDim.x + threadIdx.x;
    if (i >= LSTM_B * 256) return;
    int m = i >> 8, n = i & 255;
    const float* a = in + (size_t)m * 512;
    const float* ww = w + (size_t)n * 512;
    float s = bias[n];
    for (int k = 0; k < 512; ++k) s += a[k] * ww[k];
    out[i] = s > 0.0f ? s : 0.0f;
}

__global__ void k_fc3(const float* __restrict__ in,   // (16 x 256)
                      const float* __restrict__ w,    // (1 x 256)
                      const float* __restrict__ bias,
                      float* __restrict__ out) {      // (16)
    int m = threadIdx.x;
    if (m >= LSTM_B) return;
    const float* a = in + (size_t)m * 256;
    float s = bias[0];
    for (int k = 0; k < 256; ++k) s += a[k] * w[k];
    out[m] = s;
}

// ---------------------------------------------------------------------------
// Launch
// ---------------------------------------------------------------------------
extern "C" void kernel_launch(void* const* d_in, const int* in_sizes, int n_in,
                              void* d_out, int out_size, void* d_ws, size_t ws_size,
                              hipStream_t stream) {
    (void)in_sizes; (void)n_in; (void)out_size; (void)ws_size;

    const float* x         = (const float*)d_in[0];
    const float* W_ih0     = (const float*)d_in[1];
    const float* W_ih_rest = (const float*)d_in[2];
    const float* W_hh      = (const float*)d_in[3];
    const float* b_ih      = (const float*)d_in[4];
    const float* b_hh      = (const float*)d_in[5];
    const float* fc1_w     = (const float*)d_in[6];
    const float* fc1_b     = (const float*)d_in[7];
    const float* fc2_w     = (const float*)d_in[8];
    const float* fc2_b     = (const float*)d_in[9];
    const float* fc3_w     = (const float*)d_in[10];
    const float* fc3_b     = (const float*)d_in[11];
    float* out = (float*)d_out;

    char* ws = (char*)d_ws;
    const size_t M = (size_t)LSTM_B * LSTM_T;                 // 4096
    size_t off = 0;
    auto take = [&](size_t bytes) { size_t o = off; off = (off + bytes + 255) & ~(size_t)255; return o; };

    float*          xg   = (float*)(ws + take(M * LSTM_G * sizeof(float)));        // 64 MB
    unsigned short* seqA = (unsigned short*)(ws + take(M * LSTM_H * 2));           // 8 MB
    unsigned short* seqB = (unsigned short*)(ws + take(M * LSTM_H * 2));           // 8 MB
    unsigned short* wihb = (unsigned short*)(ws + take((size_t)LSTM_G * LSTM_H * 2)); // 8 MB
    unsigned short* whhb = (unsigned short*)(ws + take((size_t)LSTM_G * LSTM_H * 2)); // 8 MB
    unsigned short* h0   = (unsigned short*)(ws + take((size_t)LSTM_B * LSTM_H * 2));
    unsigned short* h1   = (unsigned short*)(ws + take((size_t)LSTM_B * LSTM_H * 2));
    float*          cst  = (float*)(ws + take((size_t)LSTM_B * LSTM_H * sizeof(float)));
    unsigned*       cnt  = (unsigned*)(ws + take(256));
    float*          f1o  = (float*)(ws + take((size_t)LSTM_B * 512 * sizeof(float)));
    float*          f2o  = (float*)(ws + take((size_t)LSTM_B * 256 * sizeof(float)));

    // x -> bf16 sequence buffer (layer 0 input, ld = IN = 512)
    k_f32_to_bf16<<<1024, 256, 0, stream>>>(x, seqA, LSTM_B * LSTM_T * LSTM_IN);

    unsigned short* seqIn  = seqA;
    unsigned short* seqOut = seqB;

    for (int l = 0; l < N_LAYERS; ++l) {
        const int Kl = (l == 0) ? LSTM_IN : LSTM_H;
        const float* wih_src = (l == 0) ? W_ih0
                                        : (W_ih_rest + (size_t)(l - 1) * LSTM_G * LSTM_H);

        k_f32_to_bf16<<<2048, 256, 0, stream>>>(wih_src, wihb, LSTM_G * Kl);
        k_f32_to_bf16<<<2048, 256, 0, stream>>>(W_hh + (size_t)l * LSTM_G * LSTM_H,
                                                whhb, LSTM_G * LSTM_H);

        // xg = seqIn @ Wih^T + bih + bhh : (4096/32)*(4096/64)=8192 wave-tiles
        k_gemm_xg<<<1024, 256, 0, stream>>>(seqIn, wihb,
                                            b_ih + (size_t)l * LSTM_G,
                                            b_hh + (size_t)l * LSTM_G,
                                            xg, (int)M, LSTM_G, Kl);

        k_init_state<<<16, 256, 0, stream>>>(h0, h1, cst, cnt);
        k_lstm_recurrent<<<8, 256, 0, stream>>>(whhb, xg, seqOut, h0, h1, cst, cnt);

        unsigned short* tmp = seqIn; seqIn = seqOut; seqOut = tmp;
    }

    // FC head on h[:, T-1, :] of the final layer (now in seqIn)
    k_fc1<<<(LSTM_B * 512 + 255) / 256, 256, 0, stream>>>(seqIn, fc1_w, fc1_b, f1o);
    k_fc2<<<(LSTM_B * 256 + 255) / 256, 256, 0, stream>>>(f1o, fc2_w, fc2_b, f2o);
    k_fc3<<<1, 32, 0, stream>>>(f2o, fc3_w, fc3_b, out);
}
